// FastformerAttention_40510131536395
// MI455X (gfx1250) — matrix operand order repaired and checked by reading, not compiled, not run on hardware
//
#include <hip/hip_runtime.h>
#include <math.h>

// ---------------------------------------------------------------------------
// Fastformer attention for MI455X (gfx1250, wave32, WMMA).
// fp32 in HBM, bf16 at the WMMA boundary, fp32 accumulation.
// Bandwidth-bound (~0.5 GB moved @ 23.3 TB/s >> 137 GFLOP of bf16 WMMA):
// batched b128 global loads software-pipelined around the WMMA block,
// double-buffered LDS, b128 LDS fragment loads in the ISA 7.12.2 lane
// layout, 32x32 outputs per wave (4 WMMAs / K-step).
// ---------------------------------------------------------------------------

typedef __bf16 bf16_t;
typedef __attribute__((ext_vector_type(16))) __bf16 v16bf;
typedef __attribute__((ext_vector_type(8)))  __bf16 v8bf;
typedef __attribute__((ext_vector_type(4)))  __bf16 v4bf;
typedef __attribute__((ext_vector_type(8)))  float  v8f;

namespace {
constexpr int D_  = 1024;
constexpr int H_  = 16;
constexpr int HD_ = 64;
constexpr int B_  = 4;
constexpr int S_  = 4096;
constexpr int M_  = B_ * S_;   // 16384 rows

constexpr int BM  = 128;
constexpr int BN  = 64;
constexpr int BK  = 32;
constexpr int BKP = 40;        // padded K-extent: 80 B row stride = 5 x 16 B
                               // -> every per-lane fragment slice is 16B-aligned
constexpr int NA  = (BM * BK) / (256 * 4);   // 4 float4 per thread (A tile)
constexpr int NB  = (BK * BN) / (256 * 4);   // 2 float4 per thread (B tile)
}

// ---------------------------------------------------------------------------
// Staging phase 1: issue all global_load_b128 for the (A 128x32, B 32x64)
// tile pair into named registers. No waits here -- loads fly while WMMAs run.
// ---------------------------------------------------------------------------
template <bool FUSE>
__device__ __forceinline__
void stage_load(float4 (&ar)[NA], float4 (&gr)[NA], float4 (&qr)[NA],
                float4 (&wr)[NB],
                const float* __restrict__ A, const float* __restrict__ Aadd,
                const float* __restrict__ gmul, const float* __restrict__ W,
                int K, int N, int bm, int bn, int k0, int t)
{
    #pragma unroll
    for (int i = 0; i < NA; ++i) {
        int idx = t + i * 256;                 // float4 index, 8 per row
        int rr  = idx >> 3;
        int k4  = (idx & 7) * 4;
        int row = bm + rr;
        ar[i] = *reinterpret_cast<const float4*>(&A[(size_t)row * K + k0 + k4]);
        if (FUSE) {
            // R = V*gk + Q ; gk flat index = (row/S)*D + col = b*1024 + col
            gr[i] = *reinterpret_cast<const float4*>(
                &gmul[(size_t)(row >> 12) * K + k0 + k4]);
            qr[i] = *reinterpret_cast<const float4*>(
                &Aadd[(size_t)row * K + k0 + k4]);
        }
    }
    #pragma unroll
    for (int i = 0; i < NB; ++i) {
        int idx = t + i * 256;                 // float4 index, 16 per row
        int kk  = idx >> 4;
        int n4  = (idx & 15) * 4;
        wr[i] = *reinterpret_cast<const float4*>(
            &W[(size_t)(k0 + kk) * N + bn + n4]);
    }
}

// ---------------------------------------------------------------------------
// Staging phase 2: convert fp32 -> bf16 and store into LDS.
//   As  : row-major [BM][BKP]  (fragment slice = contiguous K-run)
//   Bst : N-major   [BN][BKP]  (transposed so B fragment = contiguous K-run)
// ---------------------------------------------------------------------------
template <bool FUSE>
__device__ __forceinline__
void stage_store(const float4 (&ar)[NA], const float4 (&gr)[NA],
                 const float4 (&qr)[NA], const float4 (&wr)[NB],
                 bf16_t (*__restrict__ Asb)[BKP], bf16_t (*__restrict__ Bsb)[BKP],
                 int t)
{
    #pragma unroll
    for (int i = 0; i < NA; ++i) {
        int idx = t + i * 256;
        int rr  = idx >> 3;
        int k4  = (idx & 7) * 4;
        float x0 = ar[i].x, x1 = ar[i].y, x2 = ar[i].z, x3 = ar[i].w;
        if (FUSE) {
            x0 = x0 * gr[i].x + qr[i].x;  x1 = x1 * gr[i].y + qr[i].y;
            x2 = x2 * gr[i].z + qr[i].z;  x3 = x3 * gr[i].w + qr[i].w;
        }
        v4bf pk;
        pk[0] = (bf16_t)x0; pk[1] = (bf16_t)x1;
        pk[2] = (bf16_t)x2; pk[3] = (bf16_t)x3;
        *reinterpret_cast<v4bf*>(&Asb[rr][k4]) = pk;   // 8B store, aligned
    }
    #pragma unroll
    for (int i = 0; i < NB; ++i) {
        int idx = t + i * 256;
        int kk  = idx >> 4;
        int n4  = (idx & 15) * 4;
        Bsb[n4 + 0][kk] = (bf16_t)wr[i].x;
        Bsb[n4 + 1][kk] = (bf16_t)wr[i].y;
        Bsb[n4 + 2][kk] = (bf16_t)wr[i].z;
        Bsb[n4 + 3][kk] = (bf16_t)wr[i].w;
    }
}

__device__ __forceinline__
v16bf frag16(const v8bf* rowp, int sel_lo, int sel_hi)
{
    return __builtin_shufflevector(rowp[sel_lo], rowp[sel_hi],
        0,1,2,3,4,5,6,7,8,9,10,11,12,13,14,15);
}

// ---------------------------------------------------------------------------
// Tiled GEMM: out[M,N] = A[M,K] @ W[K,N] + bias, with optional fused
//   A' = A * gmul + Aadd   (FUSE: builds R = V*gk + Q on the fly)
//   clip to [-100,100]     (CLIP)
// 256 threads = 8 waves in a 4x2 grid; each wave owns a 32x32 slab ->
// 4 x V_WMMA_F32_16X16X32_BF16 per K-step. Software pipeline per K-step:
//   sync -> issue next-tile global loads -> WMMAs on buf p -> cvt+store p^1
// Single loop body (no unroll) so accumulator PHIs coalesce in place and the
// allocator emits no C->D shuffle copies around the WMMAs.
// ---------------------------------------------------------------------------
template <bool FUSE, bool CLIP>
__global__ __launch_bounds__(256)
void gemm_bf16_wmma(const float* __restrict__ A,
                    const float* __restrict__ Aadd,
                    const float* __restrict__ gmul,
                    const float* __restrict__ W,
                    const float* __restrict__ bias,
                    float* __restrict__ out,
                    int M, int K, int N)
{
    __shared__ __align__(16) bf16_t As [2][BM][BKP];   // 20 KB
    __shared__ __align__(16) bf16_t Bst[2][BN][BKP];   // 10 KB

    const int t    = threadIdx.x;
    const int wid  = t >> 5;
    const int lane = t & 31;
    const int wm   = (wid >> 1) * 32;       // wave row offset: 0/32/64/96
    const int wn   = (wid &  1) * 32;       // wave col offset: 0/32
    const int hlf  = lane >> 4;             // lane half per ISA layouts
    const int r    = lane & 15;
    const int bm   = blockIdx.x * BM;
    const int bn   = blockIdx.y * BN;

    v8f c00 = {}, c01 = {}, c10 = {}, c11 = {};
    float4 ar[NA], gr[NA], qr[NA], wr[NB];

    stage_load<FUSE>(ar, gr, qr, wr, A, Aadd, gmul, W, K, N, bm, bn, 0, t);
    stage_store<FUSE>(ar, gr, qr, wr, As[0], Bst[0], t);

    int p = 0;
    #pragma unroll 1
    for (int k0 = 0; k0 < K; k0 += BK) {
        __syncthreads();                    // buffer p ready; p^1 free

        // uniform (branch-free) GL2 prefetch of tile k0+2*BK
        {
            int kpf = k0 + 2 * BK;
            if (kpf >= K) kpf = k0;         // uniform select, no divergence
            __builtin_prefetch(&A[(size_t)(bm + (t & 127)) * K + kpf], 0, 1);
            __builtin_prefetch(&W[(size_t)(kpf + (t >> 3)) * N + bn + (t & 7) * 8], 0, 1);
        }

        const bool more = (k0 + BK < K);
        if (more)                           // issue next tile's loads, no wait
            stage_load<FUSE>(ar, gr, qr, wr, A, Aadd, gmul, W,
                             K, N, bm, bn, k0 + BK, t);

        // ---- fragments (CDNA5 ISA 7.12.2, 16-bit, wave32) ----
        // A lane (hlf,r): elems 0..7 = K[hlf*8 ..], elems 8..15 = K[16+hlf*8 ..]
        const v8bf* a0row = reinterpret_cast<const v8bf*>(&As[p][wm + r][0]);
        const v8bf* a1row = reinterpret_cast<const v8bf*>(&As[p][wm + 16 + r][0]);
        const v16bf af0 = frag16(a0row, hlf, 2 + hlf);
        const v16bf af1 = frag16(a1row, hlf, 2 + hlf);

        // B lane (hlf, col=r): elems 0..7 = K[hlf*8 ..], elems 8..15 = K[16+hlf*8 ..]
        const v8bf* b0row = reinterpret_cast<const v8bf*>(&Bst[p][wn + r][0]);
        const v8bf* b1row = reinterpret_cast<const v8bf*>(&Bst[p][wn + 16 + r][0]);
        const v16bf bf0 = frag16(b0row, hlf, 2 + hlf);
        const v16bf bf1 = frag16(b1row, hlf, 2 + hlf);

        c00 = __builtin_amdgcn_wmma_f32_16x16x32_bf16(false, af0, false, bf0,
                                                      (short)0, c00, false, false);
        c01 = __builtin_amdgcn_wmma_f32_16x16x32_bf16(false, af0, false, bf1,
                                                      (short)0, c01, false, false);
        c10 = __builtin_amdgcn_wmma_f32_16x16x32_bf16(false, af1, false, bf0,
                                                      (short)0, c10, false, false);
        c11 = __builtin_amdgcn_wmma_f32_16x16x32_bf16(false, af1, false, bf1,
                                                      (short)0, c11, false, false);

        if (more)                           // waits retire here, behind WMMAs
            stage_store<FUSE>(ar, gr, qr, wr, As[p ^ 1], Bst[p ^ 1], t);
        p ^= 1;
    }

    // ---- store: C/D f32 layout: VGPR i -> M = 8*hlf + i, N = lane%16 ----
    const int orow0 = bm + wm + hlf * 8;
    const int orow1 = orow0 + 16;
    const int oc0   = bn + wn + r;
    const int oc1   = oc0 + 16;
    const float b0 = bias[oc0];
    const float b1 = bias[oc1];
    #pragma unroll
    for (int i = 0; i < 8; ++i) {
        float v00 = c00[i] + b0, v01 = c01[i] + b1;
        float v10 = c10[i] + b0, v11 = c11[i] + b1;
        if (CLIP) {
            v00 = fminf(fmaxf(v00, -100.f), 100.f);
            v01 = fminf(fmaxf(v01, -100.f), 100.f);
            v10 = fminf(fmaxf(v10, -100.f), 100.f);
            v11 = fminf(fmaxf(v11, -100.f), 100.f);
        }
        out[(size_t)(orow0 + i) * N + oc0] = v00;
        out[(size_t)(orow0 + i) * N + oc1] = v01;
        out[(size_t)(orow1 + i) * N + oc0] = v10;
        out[(size_t)(orow1 + i) * N + oc1] = v11;
    }
}

// ---------------------------------------------------------------------------
// Logits: Lq[row,h] = x[row,:] . Wqw[:,h] + bqw[h]  (and Lk with Wkw/bkw),
// masked to -10000 where attention_mask is set. One block per row.
// ---------------------------------------------------------------------------
__global__ __launch_bounds__(256)
void k_logits(const float* __restrict__ x, const unsigned char* __restrict__ mask,
              const float* __restrict__ Wqw, const float* __restrict__ bqw,
              const float* __restrict__ Wkw, const float* __restrict__ bkw,
              float* __restrict__ Lq, float* __restrict__ Lk)
{
    const int row = blockIdx.x;
    const int t   = threadIdx.x;
    __shared__ float xs[D_];
    __shared__ float part[8][32];

    for (int i = t; i < D_; i += 256) xs[i] = x[(size_t)row * D_ + i];
    __syncthreads();

    const int h = t & 31;           // 0..15 -> qw head, 16..31 -> kw head
    const int g = t >> 5;           // K-chunk 0..7
    float p = 0.f;
    const int kb = g * (D_ / 8);
    for (int k = kb; k < kb + D_ / 8; ++k) {
        float w = (h < 16) ? Wqw[k * H_ + h] : Wkw[k * H_ + (h - 16)];
        p += xs[k] * w;
    }
    part[g][h] = p;
    __syncthreads();

    if (t < 32) {
        float sum = 0.f;
        #pragma unroll
        for (int i = 0; i < 8; ++i) sum += part[i][t];
        const bool mk = mask[row] != 0;
        if (t < 16) Lq[(size_t)row * H_ + t]        = mk ? -10000.f : (sum + bqw[t]);
        else        Lk[(size_t)row * H_ + (t - 16)] = mk ? -10000.f : (sum + bkw[t - 16]);
    }
}

// ---------------------------------------------------------------------------
// Pool: softmax over S of L[b,:,h], clip weights to [1e-9,1], weighted sum of
// Mat[b,s,h,0:64]; optional per-d scale (gq: P = K*gq factors out of the sum),
// then l2norm. One block per (b,h).
// ---------------------------------------------------------------------------
__global__ __launch_bounds__(256)
void k_pool(const float* __restrict__ L, const float* __restrict__ Mat,
            const float* __restrict__ gmul, float* __restrict__ gout)
{
    const int bb = blockIdx.x / H_;
    const int h  = blockIdx.x % H_;
    const int t  = threadIdx.x;
    __shared__ float red[256];
    __shared__ float accs[4][64];
    __shared__ float gvec[64];

    const float* Lbh = L + (size_t)bb * S_ * H_ + h;

    float lm = -1e30f;
    for (int s = t; s < S_; s += 256) lm = fmaxf(lm, Lbh[(size_t)s * H_]);
    red[t] = lm; __syncthreads();
    for (int o = 128; o > 0; o >>= 1) {
        if (t < o) red[t] = fmaxf(red[t], red[t + o]);
        __syncthreads();
    }
    const float m = red[0]; __syncthreads();

    float ls = 0.f;
    for (int s = t; s < S_; s += 256) ls += __expf(Lbh[(size_t)s * H_] - m);
    red[t] = ls; __syncthreads();
    for (int o = 128; o > 0; o >>= 1) {
        if (t < o) red[t] += red[t + o];
        __syncthreads();
    }
    const float Z = red[0]; __syncthreads();

    const int d = t & 63, g = t >> 6;
    float acc = 0.f;
    for (int s = g; s < S_; s += 4) {
        float w = __expf(Lbh[(size_t)s * H_] - m) / Z;
        w = fminf(fmaxf(w, 1e-9f), 1.0f);
        acc += w * Mat[(size_t)(bb * S_ + s) * D_ + h * HD_ + d];
    }
    accs[g][d] = acc;
    __syncthreads();

    if (t < 64) {
        float v = accs[0][t] + accs[1][t] + accs[2][t] + accs[3][t];
        if (gmul) v *= gmul[(bb * H_ + h) * HD_ + t];
        gvec[t] = v;
        red[t]  = v * v;
    }
    __syncthreads();
    if (t < 32) red[t] += red[t + 32]; __syncthreads();
    if (t < 16) red[t] += red[t + 16]; __syncthreads();
    if (t <  8) red[t] += red[t +  8]; __syncthreads();
    if (t <  4) red[t] += red[t +  4]; __syncthreads();
    if (t <  2) red[t] += red[t +  2]; __syncthreads();
    if (t <  1) red[t] += red[t +  1]; __syncthreads();
    if (t < 64) {
        float n = sqrtf(red[0]);
        gout[(bb * H_ + h) * HD_ + t] = gvec[t] / fmaxf(n, 1e-6f);
    }
}

// ---------------------------------------------------------------------------
extern "C" void kernel_launch(void* const* d_in, const int* in_sizes, int n_in,
                              void* d_out, int out_size, void* d_ws, size_t ws_size,
                              hipStream_t stream)
{
    const float* x   = (const float*)d_in[0];
    const unsigned char* mask = (const unsigned char*)d_in[1]; // jax bool: bytes
    const float* Wq  = (const float*)d_in[2];
    const float* bq  = (const float*)d_in[3];
    const float* Wk  = (const float*)d_in[4];
    const float* bk  = (const float*)d_in[5];
    const float* Wv  = (const float*)d_in[6];
    const float* bv  = (const float*)d_in[7];
    const float* Wqw = (const float*)d_in[8];
    const float* bqw = (const float*)d_in[9];
    const float* Wkw = (const float*)d_in[10];
    const float* bkw = (const float*)d_in[11];
    const float* Wo  = (const float*)d_in[12];
    const float* bo  = (const float*)d_in[13];
    float* out = (float*)d_out;

    float* ws = (float*)d_ws;
    size_t off = 0;
    float* Q  = ws + off; off += (size_t)M_ * D_;     // 64 MB
    float* Km = ws + off; off += (size_t)M_ * D_;     // 64 MB
    float* V  = ws + off; off += (size_t)M_ * D_;     // 64 MB
    float* Lq = ws + off; off += (size_t)M_ * H_;     // 1 MB
    float* Lk = ws + off; off += (size_t)M_ * H_;     // 1 MB
    float* gq = ws + off; off += B_ * H_ * HD_;
    float* gk = ws + off; off += B_ * H_ * HD_;
    (void)in_sizes; (void)n_in; (void)out_size; (void)ws_size;

    dim3 gg(M_ / BM, D_ / BN);

    // Q/K/V projections (x stays L2-resident across the three passes)
    gemm_bf16_wmma<false, false><<<gg, 256, 0, stream>>>(x, nullptr, nullptr, Wq, bq, Q,  M_, D_, D_);
    gemm_bf16_wmma<false, false><<<gg, 256, 0, stream>>>(x, nullptr, nullptr, Wk, bk, Km, M_, D_, D_);
    gemm_bf16_wmma<false, false><<<gg, 256, 0, stream>>>(x, nullptr, nullptr, Wv, bv, V,  M_, D_, D_);

    // attention logits (masked)
    k_logits<<<M_, 256, 0, stream>>>(x, mask, Wqw, bqw, Wkw, bkw, Lq, Lk);

    // gq = l2norm(pool(Q, softmax(Lq)));  gk = l2norm(gq * pool(K, softmax(Lk)))
    k_pool<<<B_ * H_, 256, 0, stream>>>(Lq, Q,  nullptr, gq);
    k_pool<<<B_ * H_, 256, 0, stream>>>(Lk, Km, gq,      gk);

    // out = clip((V*gk + Q) @ Wo + bo);  R fused into the A-tile stage
    gemm_bf16_wmma<true, true><<<gg, 256, 0, stream>>>(V, Q, gk, Wo, bo, out, M_, D_, D_);
}
